// NonLocal3D_67671504716358
// MI455X (gfx1250) — matrix operand order repaired and checked by reading, not compiled, not run on hardware
//
#include <hip/hip_runtime.h>
#include <hip/hip_bf16.h>

typedef __bf16 bf16;
typedef __attribute__((ext_vector_type(16))) __bf16        v16bf;
typedef __attribute__((ext_vector_type(8)))  float         v8f;
typedef __attribute__((ext_vector_type(4)))  unsigned int  u32x4;   // builtin 16B vector

// ---------------- geometry ----------------
#define BATCH 2
#define NPOS  6272          // T*H*W = 8*28*28
#define CIN   256
#define CI    128
#define ROWS  (BATCH*NPOS)  // 12544
#define ROWTILES (ROWS/16)  // 784
#define TILES_PER_B (NPOS/16) // 392

#define PHI_PITCH 136       // 32 rows x 128 bf16, padded: 272B = 68 banks -> 4-bank lane stride
#define G_PITCH   40        // 128 rows x 32 bf16, padded: 80B = 20 banks  -> conflict-free

// -------- forced-global address space helpers (avoid flat_load / DScnt) -----
typedef __attribute__((address_space(1))) const u32x4 GC4;
typedef __attribute__((address_space(1))) bf16        Gbf;
typedef __attribute__((address_space(1))) const float GCf;
typedef __attribute__((address_space(1))) float       Gf;

__device__ __forceinline__ u32x4 gload4(const void* p) {
  return *(GC4*)(unsigned long long)p;
}
__device__ __forceinline__ void gstore_bf(bf16* p, bf16 v) {
  *(Gbf*)(unsigned long long)p = v;
}
__device__ __forceinline__ float gload_f(const float* p) {
  return *(GCf*)(unsigned long long)p;
}
__device__ __forceinline__ void gstore_f(float* p, float v) {
  *(Gf*)(unsigned long long)p = v;
}

union B32U { v16bf v; u32x4 q[2]; };

__device__ __forceinline__ v8f zero8() { v8f z = {0.f,0.f,0.f,0.f,0.f,0.f,0.f,0.f}; return z; }

// A-matrix (16x32 bf16) fragment: elems e -> K = kbase + 16*hi + e
__device__ __forceinline__ v16bf load_A(const bf16* rowptr, int kbase, int hi) {
  B32U u;
  u.q[0] = gload4(rowptr + kbase + 16*hi);
  u.q[1] = gload4(rowptr + kbase + 16*hi + 8);
  return u.v;
}
// B-matrix (32x16 bf16) fragment from global: elems e -> K = kbase + 16*hi + e
__device__ __forceinline__ v16bf load_B(const bf16* rowptr, int kbase, int hi) {
  B32U u;
  const bf16* p = rowptr + kbase + 16*hi;
  u.q[0] = gload4(p);
  u.q[1] = gload4(p + 8);
  return u.v;
}
// fragment from LDS: 16 contiguous bf16 (2 x ds_load_b128)
__device__ __forceinline__ v16bf lds_frag(const bf16* p) {
  B32U u;
  u.q[0] = *(const u32x4*)p;
  u.q[1] = *(const u32x4*)(p + 8);
  return u.v;
}
__device__ __forceinline__ v8f wmma_bf(v16bf a, v16bf b, v8f c) {
  return __builtin_amdgcn_wmma_f32_16x16x32_bf16(false, a, false, b, (short)0, c, false, false);
}
// async global->LDS copy, 16B per lane (ASYNCcnt)
__device__ __forceinline__ void async_copy16(void* lds_dst, const bf16* gsrc) {
  unsigned ldsOff = (unsigned)(unsigned long long)lds_dst;
  unsigned long long ga = (unsigned long long)gsrc;
  asm volatile("global_load_async_to_lds_b128 %0, %1, off"
               :: "v"(ldsOff), "v"(ga) : "memory");
}
__device__ __forceinline__ void wait_async0() {
  asm volatile("s_wait_asynccnt 0x0" ::: "memory");
}
// reductions within each 16-lane half (rows of the C-fragment live across 16 lanes)
__device__ __forceinline__ float halfmax(float v) {
  #pragma unroll
  for (int m = 1; m <= 8; m <<= 1) v = fmaxf(v, __shfl_xor(v, m, 32));
  return v;
}
__device__ __forceinline__ float halfsum(float v) {
  #pragma unroll
  for (int m = 1; m <= 8; m <<= 1) v += __shfl_xor(v, m, 32);
  return v;
}

// ---------------- kernel 0: conversions + BN folding ----------------
__global__ void nl_prep(const float* __restrict__ x,
                        const float* __restrict__ Wg, const float* __restrict__ Wt,
                        const float* __restrict__ Wp, const float* __restrict__ Wo,
                        const float* __restrict__ bo, const float* __restrict__ gamma,
                        const float* __restrict__ beta, const float* __restrict__ mmean,
                        const float* __restrict__ mvar,
                        bf16* __restrict__ xbf, bf16* __restrict__ WgT, bf16* __restrict__ WtT,
                        bf16* __restrict__ WpT, bf16* __restrict__ WoT,
                        float* __restrict__ sc, float* __restrict__ sh2) {
  const int NX = ROWS*CIN;
  const int NW = CIN*CI;
  const int total = NX + 4*NW + CIN;
  for (int i = blockIdx.x*blockDim.x + threadIdx.x; i < total; i += gridDim.x*blockDim.x) {
    if (i < NX) { gstore_bf(xbf + i, (bf16)gload_f(x + i)); }
    else if (i < NX + 3*NW) {
      int j = (i - NX) % NW, mtx = (i - NX) / NW;
      int ci = j >> 8, c = j & 255;                 // WT[ci][c] = W[c][ci]
      const float* W = (mtx == 0) ? Wg : (mtx == 1) ? Wt : Wp;
      bf16* WT = (mtx == 0) ? WgT : (mtx == 1) ? WtT : WpT;
      gstore_bf(WT + j, (bf16)gload_f(W + c*CI + ci));
    } else if (i < NX + 4*NW) {
      int j = i - NX - 3*NW;
      int c = j >> 7, ci = j & 127;                 // WoT[c][ci] = Wo[ci][c]
      gstore_bf(WoT + j, (bf16)gload_f(Wo + ci*CIN + c));
    } else {
      int c = i - NX - 4*NW;
      float s = gload_f(gamma + c)*rsqrtf(gload_f(mvar + c) + 1e-3f);
      gstore_f(sc + c, s);
      gstore_f(sh2 + c, gload_f(beta + c) + s*(gload_f(bo + c) - gload_f(mmean + c)));
    }
  }
}

// ---------------- kernel 1: theta/phi/g projections (WMMA bf16) ----------------
__global__ void __launch_bounds__(128) nl_proj(const bf16* __restrict__ xbf,
                        const bf16* __restrict__ WtT, const bf16* __restrict__ WpT,
                        const bf16* __restrict__ WgT,
                        const float* __restrict__ bt, const float* __restrict__ bp,
                        const float* __restrict__ bg,
                        bf16* __restrict__ theta, bf16* __restrict__ phi, bf16* __restrict__ gT) {
  const int wave = threadIdx.x >> 5, lane = threadIdx.x & 31;
  const int gid = blockIdx.x*4 + wave;          // 0..783
  if (gid >= ROWTILES) return;
  const int row0 = gid*16;
  const int m = lane & 15, hi = lane >> 4;
  const int b = row0 / NPOS, n0 = row0 % NPOS;

  const bf16* xrow = xbf + (long)(row0 + m)*CIN;
  v16bf xa[8];
  #pragma unroll
  for (int kk = 0; kk < 8; ++kk) xa[kk] = load_A(xrow, 32*kk, hi);

  const bf16* Ws[3]  = {WtT, WpT, WgT};
  const float* bs[3] = {bt, bp, bg};
  #pragma unroll
  for (int mtx = 0; mtx < 3; ++mtx) {
    for (int chunk = 0; chunk < 8; ++chunk) {
      const int ci = chunk*16 + m;              // N index of C-fragment
      const bf16* wrow = Ws[mtx] + (long)ci*CIN;
      v16bf wb[8];
      #pragma unroll
      for (int kk = 0; kk < 8; ++kk) wb[kk] = load_B(wrow, 32*kk, hi);
      v8f acc = zero8();
      #pragma unroll
      for (int kk = 0; kk < 8; ++kk) acc = wmma_bf(xa[kk], wb[kk], acc);
      const float bias = gload_f(bs[mtx] + ci);
      #pragma unroll
      for (int r = 0; r < 8; ++r) {
        const float v = acc[r] + bias;
        const int rr = r + 8*hi;                // M index
        if (mtx == 0)      gstore_bf(theta + (long)(row0 + rr)*CI + ci, (bf16)v);
        else if (mtx == 1) gstore_bf(phi   + (long)(row0 + rr)*CI + ci, (bf16)v);
        else               gstore_bf(gT + ((long)b*CI + ci)*NPOS + (n0 + rr), (bf16)v);
      }
    }
  }
}

// ---------------- kernel 2: flash attention y = softmax(theta phi^T) g ----------------
// 4 waves/block share async-staged phi/g tiles in LDS (double-buffered).
__global__ void __launch_bounds__(128) nl_attn(const bf16* __restrict__ theta,
                        const bf16* __restrict__ phi, const bf16* __restrict__ gT,
                        bf16* __restrict__ ybf) {
  __shared__ __align__(16) bf16 ldsPhi[2][32*PHI_PITCH];   // 2 x 8.5KB
  __shared__ __align__(16) bf16 ldsG  [2][128*G_PITCH];    // 2 x 10KB
  __shared__ __align__(16) bf16 ldsP  [4][16*32];          // per-wave P staging
  const int tid  = threadIdx.x;
  const int wave = tid >> 5, lane = tid & 31;
  const int gid = blockIdx.x*4 + wave;            // grid is exact: no guard (barriers)
  const int b  = gid / TILES_PER_B;               // uniform across the block
  const int n0 = (gid % TILES_PER_B)*16;
  const int m = lane & 15, hi = lane >> 4;

  const bf16* thRow   = theta + ((long)b*NPOS + n0 + m)*CI;
  const bf16* phiBase = phi   + (long)b*NPOS*CI;
  const bf16* gBase   = gT    + (long)b*CI*NPOS;
  bf16* myP = ldsP[wave];

  const int prow = tid >> 2, psub = tid & 3;      // phi copy: 4 threads per row

  auto issue_copy = [&](int j, int s) {
    // phi rows j..j+31 (128 bf16 each) -> ldsPhi[s] (padded pitch)
    const bf16* src = phiBase + (long)(j + prow)*CI + psub*32;
    bf16* dst = &ldsPhi[s][prow*PHI_PITCH + psub*32];
    #pragma unroll
    for (int i = 0; i < 4; ++i) async_copy16(dst + i*8, src + i*8);
    // gT rows 0..127, cols j..j+31 -> ldsG[s] (padded pitch)
    const bf16* src2 = gBase + (long)tid*NPOS + j;
    bf16* dst2 = &ldsG[s][tid*G_PITCH];
    #pragma unroll
    for (int i = 0; i < 4; ++i) async_copy16(dst2 + i*8, src2 + i*8);
  };

  v16bf ta[4];
  #pragma unroll
  for (int kk = 0; kk < 4; ++kk) ta[kk] = load_A(thRow, 32*kk, hi);

  v8f O[8];
  #pragma unroll
  for (int n = 0; n < 8; ++n) O[n] = zero8();
  float rmax[8], rsum[8];
  #pragma unroll
  for (int r = 0; r < 8; ++r) { rmax[r] = -3.0e38f; rsum[r] = 0.f; }

  issue_copy(0, 0);
  wait_async0();
  __syncthreads();

  for (int j = 0; j < NPOS; j += 32) {
    const int cur = (j >> 5) & 1;
    if (j + 32 < NPOS) issue_copy(j + 32, cur ^ 1);   // overlap copy with compute

    const bf16* phiT = ldsPhi[cur];
    const bf16* gTl  = ldsG[cur];
    // S = theta * phi^T  (two 16x16 tiles over 32 columns)
    v16bf pb[8];
    #pragma unroll
    for (int kk = 0; kk < 4; ++kk) {
      pb[2*kk]   = lds_frag(phiT + (m     )*PHI_PITCH + 32*kk + 16*hi);
      pb[2*kk+1] = lds_frag(phiT + (m + 16)*PHI_PITCH + 32*kk + 16*hi);
    }
    v8f s0 = zero8(), s1 = zero8();
    #pragma unroll
    for (int kk = 0; kk < 4; ++kk) {
      s0 = wmma_bf(ta[kk], pb[2*kk],   s0);
      s1 = wmma_bf(ta[kk], pb[2*kk+1], s1);
    }
    // online softmax over the 32 fresh columns
    float scale[8];
    #pragma unroll
    for (int r = 0; r < 8; ++r) {
      const float mx   = halfmax(fmaxf(s0[r], s1[r]));
      const float mnew = fmaxf(rmax[r], mx);
      const float scl  = __expf(rmax[r] - mnew);
      const float p0   = __expf(s0[r] - mnew);
      const float p1   = __expf(s1[r] - mnew);
      rsum[r] = rsum[r]*scl + halfsum(p0 + p1);
      rmax[r] = mnew; scale[r] = scl;
      const int rr = r + 8*hi;                  // C-frag (M,N) -> LDS row-major P[16][32]
      myP[rr*32 + m]      = (bf16)p0;
      myP[rr*32 + 16 + m] = (bf16)p1;
    }
    #pragma unroll
    for (int n = 0; n < 8; ++n)
      #pragma unroll
      for (int r = 0; r < 8; ++r) O[n][r] *= scale[r];
    // reload P with A-fragment layout (cross-lane transpose via LDS)
    B32U u;
    u.q[0] = *(const u32x4*)(myP + m*32 + 16*hi);
    u.q[1] = *(const u32x4*)(myP + m*32 + 16*hi + 8);
    const v16bf pa = u.v;
    // O += P * g  (8 chunks of 16 Ci channels)
    v16bf gb[8];
    #pragma unroll
    for (int n = 0; n < 8; ++n)
      gb[n] = lds_frag(gTl + (16*n + m)*G_PITCH + 16*hi);
    #pragma unroll
    for (int n = 0; n < 8; ++n)
      O[n] = wmma_bf(pa, gb[n], O[n]);

    wait_async0();        // next tile landed
    __syncthreads();      // everyone done reading cur, safe to overwrite next iter
  }
  #pragma unroll
  for (int r = 0; r < 8; ++r) rsum[r] = 1.0f / rsum[r];
  #pragma unroll
  for (int n = 0; n < 8; ++n)
    #pragma unroll
    for (int r = 0; r < 8; ++r)
      gstore_bf(ybf + ((long)b*NPOS + n0 + r + 8*hi)*CI + 16*n + m,
                (bf16)(O[n][r]*rsum[r]));
}

// ---------------- kernel 3: out = x + BN(y Wo + bo) ----------------
__global__ void __launch_bounds__(128) nl_out(const bf16* __restrict__ ybf,
                        const bf16* __restrict__ WoT, const float* __restrict__ x,
                        const float* __restrict__ sc, const float* __restrict__ sh2,
                        float* __restrict__ out) {
  const int wave = threadIdx.x >> 5, lane = threadIdx.x & 31;
  const int gid = blockIdx.x*4 + wave;
  if (gid >= ROWTILES) return;
  const int row0 = gid*16;
  const int m = lane & 15, hi = lane >> 4;

  const bf16* yrow = ybf + (long)(row0 + m)*CI;
  v16bf ya[4];
  #pragma unroll
  for (int kk = 0; kk < 4; ++kk) ya[kk] = load_A(yrow, 32*kk, hi);

  for (int n = 0; n < 16; ++n) {               // 16 chunks of 16 output channels
    const int c = n*16 + m;
    const bf16* wrow = WoT + (long)c*CI;
    v16bf wb[4];
    #pragma unroll
    for (int kk = 0; kk < 4; ++kk) wb[kk] = load_B(wrow, 32*kk, hi);
    v8f acc = zero8();
    #pragma unroll
    for (int kk = 0; kk < 4; ++kk) acc = wmma_bf(ya[kk], wb[kk], acc);
    const float scc = gload_f(sc + c), shc = gload_f(sh2 + c);
    #pragma unroll
    for (int r = 0; r < 8; ++r) {
      const long row = row0 + r + 8*hi;
      gstore_f(out + row*CIN + c, gload_f(x + row*CIN + c) + scc*acc[r] + shc);
    }
  }
}

// ---------------- host launcher ----------------
extern "C" void kernel_launch(void* const* d_in, const int* in_sizes, int n_in,
                              void* d_out, int out_size, void* d_ws, size_t ws_size,
                              hipStream_t stream) {
  const float* x     = (const float*)d_in[0];
  const float* Wg    = (const float*)d_in[1];
  const float* bg    = (const float*)d_in[2];
  const float* Wt    = (const float*)d_in[3];
  const float* bt    = (const float*)d_in[4];
  const float* Wp    = (const float*)d_in[5];
  const float* bp    = (const float*)d_in[6];
  const float* Wo    = (const float*)d_in[7];
  const float* bo    = (const float*)d_in[8];
  const float* gamma = (const float*)d_in[9];
  const float* beta  = (const float*)d_in[10];
  const float* mmean = (const float*)d_in[11];
  const float* mvar  = (const float*)d_in[12];
  float* out = (float*)d_out;

  char* ws = (char*)d_ws;
  size_t off = 0;
  auto carve = [&](size_t bytes) { char* p = ws + off; off += (bytes + 255) & ~size_t(255); return p; };
  bf16* xbf   = (bf16*)carve((size_t)ROWS*CIN*2);
  bf16* WgT   = (bf16*)carve((size_t)CIN*CI*2);
  bf16* WtT   = (bf16*)carve((size_t)CIN*CI*2);
  bf16* WpT   = (bf16*)carve((size_t)CIN*CI*2);
  bf16* WoT   = (bf16*)carve((size_t)CIN*CI*2);
  float* sc   = (float*)carve(CIN*4);
  float* sh2  = (float*)carve(CIN*4);
  bf16* theta = (bf16*)carve((size_t)ROWS*CI*2);
  bf16* phi   = (bf16*)carve((size_t)ROWS*CI*2);
  bf16* gT    = (bf16*)carve((size_t)ROWS*CI*2);
  bf16* ybf   = (bf16*)carve((size_t)ROWS*CI*2);

  {
    const int total = ROWS*CIN + 4*CIN*CI + CIN;
    const int blk = 256;
    int grid = (total + blk - 1)/blk;
    if (grid > 8192) grid = 8192;
    nl_prep<<<grid, blk, 0, stream>>>(x, Wg, Wt, Wp, Wo, bo, gamma, beta, mmean, mvar,
                                      xbf, WgT, WtT, WpT, WoT, sc, sh2);
  }
  nl_proj<<<ROWTILES/4, 128, 0, stream>>>(xbf, WtT, WpT, WgT, bt, bp, bg, theta, phi, gT);
  nl_attn<<<ROWTILES/4, 128, 0, stream>>>(theta, phi, gT, ybf);
  nl_out <<<ROWTILES/4, 128, 0, stream>>>(ybf, WoT, x, sc, sh2, out);
}
